// HPLC_50714973831894
// MI455X (gfx1250) — matrix operand order repaired and checked by reading, not compiled, not run on hardware
//
#include <hip/hip_runtime.h>
#include <hip/hip_bf16.h>
#include <cstdint>
#include <cstddef>

// ---- problem constants (match reference) ----
#define RAWF   128
#define PDIM   8
#define LCOM   16
#define INC    144   // RAW + L
#define HIDC   128
#define OUTC   64

typedef __bf16 bf16_t;
typedef __attribute__((ext_vector_type(16))) __bf16 v16bf;
typedef __attribute__((ext_vector_type(8)))  __bf16 v8bf;
typedef __attribute__((ext_vector_type(8)))  float  v8f;

union ABu { v16bf v; v8bf h2[2]; };

// ---------------------------------------------------------------------------
// Weight pre-swizzle: fp32 [batch][K][Nc] -> bf16 WMMA B-fragments.
// Fragment block for (kstep kt, col-tile j) is 512 bf16: index = lane*16 + e,
// holding B[K = kt*32 + 16*(lane/16) + e][N = 16*j + lane%16].
// Zero-padded in K (k >= Kdim) and in N (col >= Nc, for padded NT).
// ---------------------------------------------------------------------------
__global__ void prep_wfrag_kernel(const float* __restrict__ W, bf16_t* __restrict__ out,
                                  int Kdim, int Ksteps, int NTpad, int Nc, int total)
{
    const int fragsz = Ksteps * NTpad * 512;
    for (int gid = blockIdx.x * blockDim.x + threadIdx.x; gid < total;
         gid += gridDim.x * blockDim.x) {
        int b    = gid / fragsz;
        int rem  = gid - b * fragsz;
        int kt   = rem / (NTpad * 512);
        int rem2 = rem - kt * (NTpad * 512);
        int j    = rem2 >> 9;
        int t    = rem2 & 511;
        int ln   = t >> 4, e = t & 15;
        int h    = ln >> 4, n = ln & 15;
        int k    = kt * 32 + 16 * h + e;
        int col  = j * 16 + n;
        float v = 0.0f;
        if (k < Kdim && col < Nc)
            v = W[((size_t)b * Kdim + k) * (size_t)Nc + (size_t)col];
        out[gid] = (bf16_t)v;
    }
}

// ---------------------------------------------------------------------------
// Build x_full[N,144] = [x | concat(pos_emb,lap_pe) @ pe_w + pe_b]
// ---------------------------------------------------------------------------
__global__ void build_x_kernel(const float* __restrict__ x, const float* __restrict__ pos,
                               const float* __restrict__ lap, const float* __restrict__ pe_w,
                               const float* __restrict__ pe_b, float* __restrict__ xf, int total)
{
    for (int gid = blockIdx.x * blockDim.x + threadIdx.x; gid < total;
         gid += gridDim.x * blockDim.x) {
        int n = gid / INC, j = gid - n * INC;
        if (j < RAWF) {
            xf[gid] = x[(size_t)n * RAWF + j];
        } else {
            int o = j - RAWF;
            float acc = pe_b[o];
            #pragma unroll
            for (int k = 0; k < PDIM; ++k)
                acc += pos[(size_t)n * PDIM + k] * pe_w[k * LCOM + o];
            #pragma unroll
            for (int k = 0; k < PDIM - 1; ++k)
                acc += lap[(size_t)n * (PDIM - 1) + k] * pe_w[(PDIM + k) * LCOM + o];
            xf[gid] = acc;
        }
    }
}

// ---------------------------------------------------------------------------
// Fused WMMA GEMM: Y[rows, Nc] = act( gather(X)[rows, Kdim] @ Wfrag + bias )
// block = 256 threads = 8 waves; 64-row tile; wave (rowTile = w&3, colHalf = w>>2)
// computes 16 rows x a contiguous run of NT/2 column tiles. NT is even so the
// per-wave tile count is a compile-time constant: no control flow around WMMA,
// EXEC stays all-ones through the entire matrix pipeline (ISA 7.12 requirement).
// ---------------------------------------------------------------------------
template<int NT, int KSTEPS>
__global__ __launch_bounds__(256) void wmma_gemm_kernel(
    const float* __restrict__ X,
    const int*   __restrict__ in_idx,    // nullable gather: row = in_idx[seg*Mseg + r]
    const bf16_t* __restrict__ Wf, int wstride,
    const float* __restrict__ bias,      // nullable: bias[seg*Nc + col]
    float*       __restrict__ Y,
    const int*   __restrict__ out_idx,   // nullable scatter
    int Mseg, int Kdim, int Nc, int act)
{
    constexpr int KPAD  = KSTEPS * 32;
    constexpr int PITCH = KPAD + 8;       // bank-conflict-free LDS pitch
    constexpr int JN    = NT / 2;         // tiles per wave (compile-time)
    __shared__ bf16_t ldsA[64 * PITCH];

    const int seg = blockIdx.y;
    const int r0  = blockIdx.x * 64;
    const int tid = threadIdx.x;
    const size_t segBase = (size_t)seg * (size_t)Mseg;

    // stage 64 x KPAD activation tile as bf16 (zero-pad K)
    for (int i = tid; i < 64 * KPAD; i += 256) {
        int r = i / KPAD, k = i - r * KPAD;
        int rg = r0 + r; if (rg >= Mseg) rg = 0;
        size_t srow = in_idx ? (size_t)in_idx[segBase + rg] : (segBase + (size_t)rg);
        float v = (k < Kdim) ? X[srow * (size_t)Kdim + k] : 0.0f;
        ldsA[r * PITCH + k] = (bf16_t)v;
    }
    __syncthreads();

    const int lane = tid & 31;
    const int wave = tid >> 5;
    const int rowTile = wave & 3;
    const int colHalf = wave >> 2;
    const int jbase   = colHalf * JN;
    const int m = lane & 15;
    const int h = lane >> 4;

    v8f acc[JN] = {};
    const bf16_t* arow = &ldsA[(rowTile * 16 + m) * PITCH];
    const bf16_t* wbase = Wf + (size_t)seg * (size_t)wstride + (size_t)lane * 16;

    #pragma unroll
    for (int kt = 0; kt < KSTEPS; ++kt) {
        // A fragment (16x32 bf16): elem e -> K = kt*32 + 16h + e
        ABu a;
        a.h2[0] = *(const v8bf*)(arow + kt * 32 + 16 * h);
        a.h2[1] = *(const v8bf*)(arow + kt * 32 + 16 * h + 8);
        #pragma unroll
        for (int jj = 0; jj < JN; ++jj) {
            ABu b;
            const bf16_t* wp = wbase + (((size_t)(kt * NT + jbase + jj)) << 9);
            b.h2[0] = *(const v8bf*)(wp);
            b.h2[1] = *(const v8bf*)(wp + 8);
            acc[jj] = __builtin_amdgcn_wmma_f32_16x16x32_bf16(
                false, a.v, false, b.v, (short)0, acc[jj], false, false);
        }
    }

    // store: C/D layout -> element v of acc = row (v + 8h), col (16*(jbase+jj) + m)
    #pragma unroll
    for (int jj = 0; jj < JN; ++jj) {
        const int col = (jbase + jj) * 16 + m;
        if (col < Nc) {
            const float bv = bias ? bias[(size_t)seg * Nc + col] : 0.0f;
            #pragma unroll
            for (int v = 0; v < 8; ++v) {
                const int rl = rowTile * 16 + 8 * h + v;
                const int rg = r0 + rl;
                if (rg < Mseg) {
                    float val = acc[jj][v] + bv;
                    if (act) val = (val > 0.0f) ? val : 0.01f * val;   // leaky_relu(0.01)
                    size_t orow = out_idx ? (size_t)out_idx[segBase + rg] : (segBase + (size_t)rg);
                    Y[orow * (size_t)Nc + col] = val;
                }
            }
        }
    }
}

// ---------------------------------------------------------------------------
// GCN aggregation helpers
// ---------------------------------------------------------------------------
__global__ void fill_one_kernel(float* __restrict__ p, int n)
{
    for (int i = blockIdx.x * blockDim.x + threadIdx.x; i < n; i += gridDim.x * blockDim.x)
        p[i] = 1.0f;
}

__global__ void deg_count_kernel(const int* __restrict__ rowi, float* __restrict__ deg, int E)
{
    for (int e = blockIdx.x * blockDim.x + threadIdx.x; e < E; e += gridDim.x * blockDim.x)
        atomicAdd(&deg[rowi[e]], 1.0f);
}

__global__ void rsqrt_kernel(float* __restrict__ p, int n)
{
    for (int i = blockIdx.x * blockDim.x + threadIdx.x; i < n; i += gridDim.x * blockDim.x)
        p[i] = rsqrtf(p[i]);
}

// out[i,f] = Y[i,f] * rs[i]^2   (self-loop term, also initializes out)
__global__ void selfloop_kernel(const float* __restrict__ Yd, float* __restrict__ out,
                                const float* __restrict__ rs, int F, int total)
{
    for (int gid = blockIdx.x * blockDim.x + threadIdx.x; gid < total;
         gid += gridDim.x * blockDim.x) {
        int i = gid / F;
        float r = rs[i];
        out[gid] = Yd[gid] * r * r;
    }
}

// out[row] += Y[col] * rs[row]*rs[col]; one wave per edge, lanes over features
__global__ void edge_agg_kernel(const float* __restrict__ Yd, float* __restrict__ out,
                                const int* __restrict__ rowi, const int* __restrict__ coli,
                                const float* __restrict__ rs, int E, int F)
{
    int gid  = blockIdx.x * blockDim.x + threadIdx.x;
    int wid  = gid >> 5, lane = gid & 31;
    int nw   = (gridDim.x * blockDim.x) >> 5;
    for (int e = wid; e < E; e += nw) {
        int r = rowi[e], c = coli[e];
        float nrm = rs[r] * rs[c];
        const float* src = Yd + (size_t)c * F;
        float*       dst = out + (size_t)r * F;
        for (int f = lane; f < F; f += 32)
            atomicAdd(&dst[f], src[f] * nrm);
    }
}

__global__ void bias_act_kernel(float* __restrict__ out, const float* __restrict__ b,
                                int F, int total, int relu)
{
    for (int gid = blockIdx.x * blockDim.x + threadIdx.x; gid < total;
         gid += gridDim.x * blockDim.x) {
        float v = out[gid] + b[gid % F];
        if (relu) v = fmaxf(v, 0.0f);
        out[gid] = v;
    }
}

// ---------------------------------------------------------------------------
static inline size_t align256(size_t x) { return (x + 255u) & ~(size_t)255u; }

extern "C" void kernel_launch(void* const* d_in, const int* in_sizes, int n_in,
                              void* d_out, int out_size, void* d_ws, size_t ws_size,
                              hipStream_t stream)
{
    (void)n_in; (void)out_size; (void)ws_size;
    const float* x     = (const float*)d_in[0];
    const float* pos   = (const float*)d_in[1];
    const float* lap   = (const float*)d_in[2];
    const int*   eidx  = (const int*)  d_in[3];
    const int*   com   = (const int*)  d_in[4];
    const float* pe_w  = (const float*)d_in[5];
    const float* pe_b  = (const float*)d_in[6];
    const float* fc1_w = (const float*)d_in[7];
    const float* fc1_b = (const float*)d_in[8];
    const float* fc2_w = (const float*)d_in[9];
    const float* fc2_b = (const float*)d_in[10];
    const float* fc3_w = (const float*)d_in[11];
    const float* fc3_b = (const float*)d_in[12];
    const float* w1    = (const float*)d_in[13];
    const float* b1    = (const float*)d_in[14];
    const float* w2    = (const float*)d_in[15];
    const float* b2    = (const float*)d_in[16];
    const float* w3    = (const float*)d_in[17];
    const float* b3    = (const float*)d_in[18];
    float* out = (float*)d_out;

    const int N = in_sizes[0] / RAWF;      // 100000
    const int E = in_sizes[3] / 2;         // 1600000
    const int S = N / LCOM;                // 6250
    const int* erow = eidx;
    const int* ecol = eidx + E;

    // workspace layout
    char* ws = (char*)d_ws;
    size_t off = 0;
    float* xf   = (float*)(ws + off); off = align256(off + (size_t)N * INC  * 4);
    float* bufA = (float*)(ws + off); off = align256(off + (size_t)N * HIDC * 4);
    float* bufB = (float*)(ws + off); off = align256(off + (size_t)N * HIDC * 4);
    float* rs   = (float*)(ws + off); off = align256(off + (size_t)N * 4);
    bf16_t* fc1f = (bf16_t*)(ws + off);
    bf16_t* fc2f = fc1f + (size_t)LCOM * 5 * 8  * 512;   // fc1: NT=8,  K=5
    bf16_t* fc3f = fc2f + (size_t)LCOM * 4 * 8  * 512;   // fc2: NT=8,  K=4
    bf16_t* w1f  = fc3f + (size_t)LCOM * 4 * 10 * 512;   // fc3: NT=10, K=4 (col-padded 144->160)
    bf16_t* w2f  = w1f + 5 * 8 * 512;
    bf16_t* w3f  = w2f + 4 * 8 * 512;

    const int TB = 256;
    auto gs = [](int total, int tb) { return (total + tb - 1) / tb; };

    // 1) weights -> bf16 WMMA fragments
    prep_wfrag_kernel<<<gs(LCOM*5*8*512,  TB), TB, 0, stream>>>(fc1_w, fc1f, INC,  5, 8,  HIDC, LCOM*5*8*512);
    prep_wfrag_kernel<<<gs(LCOM*4*8*512,  TB), TB, 0, stream>>>(fc2_w, fc2f, HIDC, 4, 8,  HIDC, LCOM*4*8*512);
    prep_wfrag_kernel<<<gs(LCOM*4*10*512, TB), TB, 0, stream>>>(fc3_w, fc3f, HIDC, 4, 10, INC,  LCOM*4*10*512);
    prep_wfrag_kernel<<<gs(5*8*512, TB), TB, 0, stream>>>(w1, w1f, INC,  5, 8, HIDC, 5*8*512);
    prep_wfrag_kernel<<<gs(4*8*512, TB), TB, 0, stream>>>(w2, w2f, HIDC, 4, 8, HIDC, 4*8*512);
    prep_wfrag_kernel<<<gs(4*4*512, TB), TB, 0, stream>>>(w3, w3f, HIDC, 4, 4, OUTC, 4*4*512);

    // 2) positional embedding + concat
    build_x_kernel<<<gs(N*INC, TB), TB, 0, stream>>>(x, pos, lap, pe_w, pe_b, xf, N*INC);

    // 3) degrees (self-loop => init 1) and deg^-1/2
    fill_one_kernel<<<gs(N, TB), TB, 0, stream>>>(rs, N);
    deg_count_kernel<<<gs(E, TB), TB, 0, stream>>>(erow, rs, E);
    rsqrt_kernel<<<gs(N, TB), TB, 0, stream>>>(rs, N);

    // 4) per-community fused 3-layer MLP (gather -> 3 WMMA GEMMs -> scatter)
    dim3 gMLP((unsigned)((S + 63) / 64), (unsigned)LCOM);
    wmma_gemm_kernel<8,5> <<<gMLP, TB, 0, stream>>>(xf,   com,     fc1f, 5*8*512,  fc1_b, bufA, nullptr, S, INC,  HIDC, 1);
    wmma_gemm_kernel<8,4> <<<gMLP, TB, 0, stream>>>(bufA, nullptr, fc2f, 4*8*512,  fc2_b, bufB, nullptr, S, HIDC, HIDC, 1);
    wmma_gemm_kernel<10,4><<<gMLP, TB, 0, stream>>>(bufB, nullptr, fc3f, 4*10*512, fc3_b, xf,   com,     S, HIDC, INC,  1);

    // 5) GCN layers: dense WMMA GEMM, then self-loop + edge scatter-add, then bias(+relu)
    dim3 gGCN((unsigned)((N + 63) / 64), 1u);
    const int EAG = 4096;  // edge-agg blocks

    // layer 1: [N,144]@w1 -> bufA; aggregate -> bufB; relu(+b1)
    wmma_gemm_kernel<8,5><<<gGCN, TB, 0, stream>>>(xf, nullptr, w1f, 0, nullptr, bufA, nullptr, N, INC, HIDC, 0);
    selfloop_kernel<<<gs(N*HIDC, TB), TB, 0, stream>>>(bufA, bufB, rs, HIDC, N*HIDC);
    edge_agg_kernel<<<EAG, TB, 0, stream>>>(bufA, bufB, erow, ecol, rs, E, HIDC);
    bias_act_kernel<<<gs(N*HIDC, TB), TB, 0, stream>>>(bufB, b1, HIDC, N*HIDC, 1);

    // layer 2
    wmma_gemm_kernel<8,4><<<gGCN, TB, 0, stream>>>(bufB, nullptr, w2f, 0, nullptr, bufA, nullptr, N, HIDC, HIDC, 0);
    selfloop_kernel<<<gs(N*HIDC, TB), TB, 0, stream>>>(bufA, bufB, rs, HIDC, N*HIDC);
    edge_agg_kernel<<<EAG, TB, 0, stream>>>(bufA, bufB, erow, ecol, rs, E, HIDC);
    bias_act_kernel<<<gs(N*HIDC, TB), TB, 0, stream>>>(bufB, b2, HIDC, N*HIDC, 1);

    // layer 3 (no relu) -> d_out
    wmma_gemm_kernel<4,4><<<gGCN, TB, 0, stream>>>(bufB, nullptr, w3f, 0, nullptr, bufA, nullptr, N, HIDC, OUTC, 0);
    selfloop_kernel<<<gs(N*OUTC, TB), TB, 0, stream>>>(bufA, out, rs, OUTC, N*OUTC);
    edge_agg_kernel<<<EAG, TB, 0, stream>>>(bufA, out, erow, ecol, rs, E, OUTC);
    bias_act_kernel<<<gs(N*OUTC, TB), TB, 0, stream>>>(out, b3, OUTC, N*OUTC, 0);
}
